// PDE_RNN_58961311039880
// MI455X (gfx1250) — matrix operand order repaired and checked by reading, not compiled, not run on hardware
//
#include <hip/hip_runtime.h>
#include <cstdint>
#include <cstddef>

// ---------------- problem constants (from reference setup_inputs) ----------
#define HID    1024
#define LAYERS 3
#define OUTN   2500
#define GRIDN  50
#define MODES  32
#define BATCH  32
#define TSTEPS 64          // time_size is a device-side scalar fixed to 64 in setup_inputs()
#define GATE3  (3*HID)     // 3072

// ---------------- CDNA5 WMMA types ----------------------------------------
typedef __attribute__((ext_vector_type(16))) __bf16       v16bf;
typedef __attribute__((ext_vector_type(8)))  float        v8f;
typedef __attribute__((ext_vector_type(8)))  unsigned int v8u;

__device__ __forceinline__ unsigned int f2bf1(float f) {
  unsigned int u = __float_as_uint(f);
  return (u + 0x7FFFu + ((u >> 16) & 1u)) >> 16;     // RNE f32 -> bf16
}

// A fragment: 16x32 bf16, contiguous k order (k = 16h + s), same as B.
// lanes<16 hold K base+0..15; lanes>=16 hold base+16..31.
__device__ __forceinline__ v16bf load_a_frag(const unsigned short* arow, int k0, int hi) {
  const int ka = k0 + (hi ? 16 : 0);
  uint4 a0 = *(const uint4*)(arow + ka);        // 8 bf16, K ka..ka+7
  uint4 a1 = *(const uint4*)(arow + ka + 8);    // 8 bf16, K ka+8..ka+15
  v8u au;
  au[0] = a0.x; au[1] = a0.y; au[2] = a0.z; au[3] = a0.w;
  au[4] = a1.x; au[5] = a1.y; au[6] = a1.z; au[7] = a1.w;
  return __builtin_bit_cast(v16bf, au);
}

// B fragment: 32x16 bf16; lanes 0-15 K=k0..k0+15, lanes 16-31 K=k0+16..k0+31.
__device__ __forceinline__ v16bf load_b_frag(const unsigned short* wrow, int k0, int hi) {
  v8u bu = *(const v8u*)(wrow + k0 + (hi ? 16 : 0));  // 32B contiguous, 32B aligned
  return __builtin_bit_cast(v16bf, bu);
}

// ---------------- one-time f32 -> bf16 weight conversion -------------------
__global__ void cvt_bf16_kernel(const float* __restrict__ src,
                                unsigned short* __restrict__ dst, int n) {
  int i = blockIdx.x * blockDim.x + threadIdx.x;
  if (i < n) dst[i] = (unsigned short)f2bf1(src[i]);
}

// ---------------- state init: xbf <- bf16(input), h <- 0, hbf <- 0 ---------
__global__ void init_state_kernel(const float* __restrict__ xin,
                                  unsigned short* __restrict__ xbf,
                                  float* __restrict__ h,
                                  unsigned short* __restrict__ hbf) {
  int i = blockIdx.x * blockDim.x + threadIdx.x;
  if (i < BATCH * HID) xbf[i] = (unsigned short)f2bf1(xin[i]);
  if (i < LAYERS * BATCH * HID) { h[i] = 0.0f; hbf[i] = 0; }
}

// ---------------- GRU gate GEMMs: gi = inp*Wih^T + bih, gh = h*Whh^T + bhh -
// grid.x = 96; 4 waves/block: wave = which(gi/gh) x mtile.
// Each wave: two 16x16 f32 tiles (N-pair) sharing the A fragment,
// K=1024 as 32x (2x V_WMMA_F32_16X16X32_BF16).
__global__ __launch_bounds__(128)
void gates_gemm_kernel(const unsigned short* __restrict__ Ainp, // [B,HID] bf16 layer input
                       const unsigned short* __restrict__ Ah,   // [B,HID] bf16 h state
                       const unsigned short* __restrict__ Wih,  // [GATE3,HID] bf16
                       const unsigned short* __restrict__ Whh,  // [GATE3,HID] bf16
                       const float* __restrict__ bih,
                       const float* __restrict__ bhh,
                       float* __restrict__ gi, float* __restrict__ gh) {
  const int lane  = threadIdx.x & 31;
  const int wave  = threadIdx.x >> 5;
  const int which = wave >> 1;          // 0 -> gi, 1 -> gh
  const int mtile = wave & 1;           // batch tile 0/1

  const unsigned short* A    = which ? Ah  : Ainp;
  const unsigned short* W    = which ? Whh : Wih;
  const float*          bias = which ? bhh : bih;
  float*                C    = which ? gh  : gi;

  const int hi    = lane >> 4;                       // half-wave id
  const int mrow  = mtile * 16 + (lane & 15);        // A row held by this lane
  const int ncol0 = blockIdx.x * 32 + (lane & 15);   // column, N-tile pair 2b,2b+1
  const int ncol1 = ncol0 + 16;
  const unsigned short* arow  = A + (size_t)mrow * HID;
  const unsigned short* wrow0 = W + (size_t)ncol0 * HID;
  const unsigned short* wrow1 = W + (size_t)ncol1 * HID;

  v8f c0 = {0.f, 0.f, 0.f, 0.f, 0.f, 0.f, 0.f, 0.f};
  v8f c1 = {0.f, 0.f, 0.f, 0.f, 0.f, 0.f, 0.f, 0.f};

  for (int k0 = 0; k0 < HID; k0 += 32) {
    v16bf a  = load_a_frag(arow, k0, hi);
    v16bf b0 = load_b_frag(wrow0, k0, hi);
    v16bf b1 = load_b_frag(wrow1, k0, hi);
    // speculative prefetch of next K-slab (silently dropped if OOB)
    __builtin_prefetch(wrow0 + k0 + 64, 0, 1);
    __builtin_prefetch(wrow1 + k0 + 64, 0, 1);
    c0 = __builtin_amdgcn_wmma_f32_16x16x32_bf16(false, a, false, b0,
                                                 (short)0, c0, false, false);
    c1 = __builtin_amdgcn_wmma_f32_16x16x32_bf16(false, a, false, b1,
                                                 (short)0, c1, false, false);
  }

  // D layout: lane group hi selects M rows 0-7 / 8-15 of each 16x16 tile.
  const float bv0   = bias[ncol0];
  const float bv1   = bias[ncol1];
  const int   mbase = mtile * 16 + hi * 8;
#pragma unroll
  for (int r = 0; r < 8; ++r) {
    C[(size_t)(mbase + r) * GATE3 + ncol0] = c0[r] + bv0;
    C[(size_t)(mbase + r) * GATE3 + ncol1] = c1[r] + bv1;
  }
}

// ---------------- GRU pointwise update (f32 state + bf16 shadow) -----------
__global__ __launch_bounds__(256)
void gru_pointwise_kernel(const float* __restrict__ gi,
                          const float* __restrict__ gh,
                          float* __restrict__ h,
                          unsigned short* __restrict__ hbf) {
  int i = blockIdx.x * blockDim.x + threadIdx.x;
  if (i >= BATCH * HID) return;
  int b = i / HID, j = i % HID;
  const float* gib = gi + (size_t)b * GATE3;
  const float* ghb = gh + (size_t)b * GATE3;
  float ir = gib[j], iz = gib[HID + j], in_ = gib[2 * HID + j];
  float hr = ghb[j], hz = ghb[HID + j], hn  = ghb[2 * HID + j];
  float r = 1.0f / (1.0f + __expf(-(ir + hr)));
  float z = 1.0f / (1.0f + __expf(-(iz + hz)));
  float n = tanhf(in_ + r * hn);
  float hn2 = (1.0f - z) * n + z * h[i];
  h[i]   = hn2;
  hbf[i] = (unsigned short)f2bf1(hn2);
}

// ---------------- FC head: out = h2*fcw^T + fcb ----------------------------
// grid.x = 79 (N-tile pairs covering 157 tiles), 2 waves/block (mtile 0/1).
__global__ __launch_bounds__(64)
void fc_gemm_kernel(const unsigned short* __restrict__ A,   // [B,HID] bf16
                    const unsigned short* __restrict__ Wfc, // [OUTN,HID] bf16
                    const float* __restrict__ bias,
                    float* __restrict__ out) {              // [B,OUTN]
  const int lane  = threadIdx.x & 31;
  const int mtile = threadIdx.x >> 5;
  const int hi    = lane >> 4;
  const int mrow  = mtile * 16 + (lane & 15);
  const int ncol0 = blockIdx.x * 32 + (lane & 15);
  const int ncol1 = ncol0 + 16;
  const int ncl0  = ncol0 < OUTN ? ncol0 : (OUTN - 1);  // keep loads in-bounds
  const int ncl1  = ncol1 < OUTN ? ncol1 : (OUTN - 1);
  const unsigned short* arow  = A + (size_t)mrow * HID;
  const unsigned short* wrow0 = Wfc + (size_t)ncl0 * HID;
  const unsigned short* wrow1 = Wfc + (size_t)ncl1 * HID;

  v8f c0 = {0.f, 0.f, 0.f, 0.f, 0.f, 0.f, 0.f, 0.f};
  v8f c1 = {0.f, 0.f, 0.f, 0.f, 0.f, 0.f, 0.f, 0.f};
  for (int k0 = 0; k0 < HID; k0 += 32) {
    v16bf a  = load_a_frag(arow, k0, hi);
    v16bf b0 = load_b_frag(wrow0, k0, hi);
    v16bf b1 = load_b_frag(wrow1, k0, hi);
    c0 = __builtin_amdgcn_wmma_f32_16x16x32_bf16(false, a, false, b0,
                                                 (short)0, c0, false, false);
    c1 = __builtin_amdgcn_wmma_f32_16x16x32_bf16(false, a, false, b1,
                                                 (short)0, c1, false, false);
  }

  const int mbase = mtile * 16 + hi * 8;
  if (ncol0 < OUTN) {
    const float bv0 = bias[ncol0];
#pragma unroll
    for (int r = 0; r < 8; ++r)
      out[(size_t)(mbase + r) * OUTN + ncol0] = c0[r] + bv0;
  }
  if (ncol1 < OUTN) {
    const float bv1 = bias[ncol1];
#pragma unroll
    for (int r = 0; r < 8; ++r)
      out[(size_t)(mbase + r) * OUTN + ncol1] = c1[r] + bv1;
  }
}

// ---------------- spectral crop: x_next = crop(fftshift(Re(FFT2(out)))) ----
// Linear separable DFT; freq index j in [0,32) maps to f = j-16.
// One block per batch element; everything staged through LDS; x emitted bf16.
__global__ __launch_bounds__(256)
void spectral_kernel(const float* __restrict__ out,   // [B,OUTN]
                     unsigned short* __restrict__ xbf) { // [B,HID] bf16
  __shared__ float sOut[OUTN];             // 50x50 grid
  __shared__ float sCos[MODES][GRIDN];     // cos(2*pi*f*n/50), f=j-16
  __shared__ float sSin[MODES][GRIDN];
  __shared__ float sGr[GRIDN][MODES];      // row-DFT real
  __shared__ float sGi[GRIDN][MODES];      // row-DFT imag

  const int b   = blockIdx.x;
  const int tid = threadIdx.x;
  const float* ob = out + (size_t)b * OUTN;

  for (int i = tid; i < OUTN; i += 256) sOut[i] = ob[i];
  for (int i = tid; i < MODES * GRIDN; i += 256) {
    int j = i / GRIDN, n = i % GRIDN;
    float f  = (float)(j - 16);
    float th = 6.283185307179586f * f * (float)n / (float)GRIDN;
    float s, c;
    __sincosf(th, &s, &c);
    sCos[j][n] = c;
    sSin[j][n] = s;
  }
  __syncthreads();

  // Stage 1: G[n1,k2] = sum_n2 out[n1,n2] * e^{-i*2pi*f(k2)*n2/50}
  for (int i = tid; i < GRIDN * MODES; i += 256) {
    int n1 = i / MODES, k2 = i % MODES;
    float gr = 0.f, gim = 0.f;
    const float* row = &sOut[n1 * GRIDN];
#pragma unroll 5
    for (int n2 = 0; n2 < GRIDN; ++n2) {
      float v = row[n2];
      gr  += v * sCos[k2][n2];
      gim -= v * sSin[k2][n2];
    }
    sGr[n1][k2] = gr;
    sGi[n1][k2] = gim;
  }
  __syncthreads();

  // Stage 2: Re F[k1,k2] = sum_n1 Gr*cos + Gi*sin -> next-step input (bf16)
  for (int i = tid; i < MODES * MODES; i += 256) {
    int k1 = i / MODES, k2 = i % MODES;
    float acc = 0.f;
#pragma unroll 5
    for (int n1 = 0; n1 < GRIDN; ++n1)
      acc += sGr[n1][k2] * sCos[k1][n1] + sGi[n1][k2] * sSin[k1][n1];
    xbf[(size_t)b * HID + i] = (unsigned short)f2bf1(acc);
  }
}

// ---------------- host orchestration ---------------------------------------
extern "C" void kernel_launch(void* const* d_in, const int* in_sizes, int n_in,
                              void* d_out, int out_size, void* d_ws, size_t ws_size,
                              hipStream_t stream) {
  const float* xin = (const float*)d_in[0];
  const float* Wih = (const float*)d_in[1];
  const float* Whh = (const float*)d_in[2];
  const float* bih = (const float*)d_in[3];
  const float* bhh = (const float*)d_in[4];
  const float* fcw = (const float*)d_in[5];
  const float* fcb = (const float*)d_in[6];
  float* out = (float*)d_out;

  // workspace carve-out (~44.5 MB total)
  char*  ws  = (char*)d_ws;
  size_t off = 0;
  auto carve = [&](size_t bytes) -> void* {
    void* p = ws + off;
    off = (off + bytes + 255) & ~(size_t)255;
    return p;
  };
  unsigned short* WihB = (unsigned short*)carve((size_t)LAYERS * GATE3 * HID * 2);
  unsigned short* WhhB = (unsigned short*)carve((size_t)LAYERS * GATE3 * HID * 2);
  unsigned short* FcwB = (unsigned short*)carve((size_t)OUTN * HID * 2);
  float*          hbuf = (float*)carve((size_t)LAYERS * BATCH * HID * 4);
  unsigned short* hbf  = (unsigned short*)carve((size_t)LAYERS * BATCH * HID * 2);
  unsigned short* xbf  = (unsigned short*)carve((size_t)BATCH * HID * 2);
  float*         gibuf = (float*)carve((size_t)BATCH * GATE3 * 4);
  float*         ghbuf = (float*)carve((size_t)BATCH * GATE3 * 4);

  // 1) one-time weight conversion to bf16 (redone every call: deterministic)
  {
    int n1 = LAYERS * GATE3 * HID;
    cvt_bf16_kernel<<<(n1 + 255) / 256, 256, 0, stream>>>(Wih, WihB, n1);
    cvt_bf16_kernel<<<(n1 + 255) / 256, 256, 0, stream>>>(Whh, WhhB, n1);
    int n2 = OUTN * HID;
    cvt_bf16_kernel<<<(n2 + 255) / 256, 256, 0, stream>>>(fcw, FcwB, n2);
  }

  // 2) state init
  {
    int n = LAYERS * BATCH * HID;
    init_state_kernel<<<(n + 255) / 256, 256, 0, stream>>>(xin, xbf, hbuf, hbf);
  }

  // 3) sequential recurrence
  for (int t = 0; t < TSTEPS; ++t) {
    for (int l = 0; l < LAYERS; ++l) {
      const unsigned short* inp =
          (l == 0) ? xbf : (hbf + (size_t)(l - 1) * BATCH * HID);
      gates_gemm_kernel<<<GATE3 / 32, 128, 0, stream>>>(
          inp, hbf + (size_t)l * BATCH * HID,
          WihB + (size_t)l * GATE3 * HID, WhhB + (size_t)l * GATE3 * HID,
          bih + (size_t)l * GATE3, bhh + (size_t)l * GATE3,
          gibuf, ghbuf);
      gru_pointwise_kernel<<<(BATCH * HID + 255) / 256, 256, 0, stream>>>(
          gibuf, ghbuf, hbuf + (size_t)l * BATCH * HID,
          hbf + (size_t)l * BATCH * HID);
    }
    float* out_t = out + (size_t)t * BATCH * OUTN;
    fc_gemm_kernel<<<(OUTN + 31) / 32, 64, 0, stream>>>(
        hbf + (size_t)2 * BATCH * HID, FcwB, fcb, out_t);
    spectral_kernel<<<BATCH, 256, 0, stream>>>(out_t, xbf);
  }
  (void)in_sizes; (void)n_in; (void)out_size; (void)ws_size;
}
